// FlashAttention_45251775431224
// MI455X (gfx1250) — matrix operand order repaired and checked by reading, not compiled, not run on hardware
//
#include <hip/hip_runtime.h>
#include <hip/hip_bf16.h>

typedef __attribute__((ext_vector_type(16))) __bf16 v16bf;
typedef __attribute__((ext_vector_type(8)))  float  v8f;
typedef __attribute__((ext_vector_type(4)))  __bf16 v4bf;

#define BATCH 4
#define SEQ   4096
#define DMODEL 2048
#define NHEAD 16
#define HDIM  128
#define MTOT  (BATCH * SEQ)   // 16384

#if defined(__has_builtin)
#if __has_builtin(__builtin_amdgcn_tensor_load_to_lds) && __has_builtin(__builtin_amdgcn_s_wait_tensorcnt)
#define USE_TDM 1
#endif
#endif
#ifndef USE_TDM
#define USE_TDM 0
#endif

// ---------------------------------------------------------------------------
// fragment loader: two contiguous 16-byte chunks -> one 16x32 / 32x16 bf16 frag
// ---------------------------------------------------------------------------
__device__ __forceinline__ v16bf load_frag2(const __bf16* p0, const __bf16* p1) {
    union { v16bf v; uint4 u[2]; } t;
    t.u[0] = *(const uint4*)p0;
    t.u[1] = *(const uint4*)p1;
    return t.v;
}

// ---------------------------------------------------------------------------
// Tensor Data Mover: issue a 2D (or 1D when tile1==1) tile load Global->LDS.
// D# built per CDNA5 ISA ch.8: group0 = {count=1, lds_addr, global_addr, type=2},
// group1 = {data_size=2B, tensor_dim0/1, tile_dim0/1, tensor_dim0_stride}.
// ---------------------------------------------------------------------------
__device__ __forceinline__ void tdm_load_2d(unsigned lds_off, const void* gsrc,
                                            unsigned tile0, unsigned tile1,
                                            unsigned ten0, unsigned ten1,
                                            unsigned long long stride0) {
#if USE_TDM
    typedef unsigned int u32x4 __attribute__((ext_vector_type(4)));
    typedef int i32x8 __attribute__((ext_vector_type(8)));
    typedef int i32x4 __attribute__((ext_vector_type(4)));
    unsigned long long ga = (unsigned long long)gsrc;
    u32x4 g0;
    g0.x = 1u;                                            // count=1 (valid D#)
    g0.y = lds_off;                                       // lds_addr
    g0.z = (unsigned)ga;                                  // global_addr[31:0]
    g0.w = (unsigned)((ga >> 32) & 0x1FFFFFFu) | (2u << 30); // addr[56:32] | type=2
    i32x8 g1;
    g1[0] = (int)(1u << 16);                              // data_size=1 -> 2 bytes
    g1[1] = (int)((ten0 & 0xFFFFu) << 16);                // tensor_dim0[15:0]
    g1[2] = (int)((ten0 >> 16) | ((ten1 & 0xFFFFu) << 16));// dim0 hi | dim1 lo
    g1[3] = (int)((ten1 >> 16) | (tile0 << 16));          // dim1 hi | tile_dim0
    g1[4] = (int)(tile1 & 0xFFFFu);                       // tile_dim1 (tile_dim2=0)
    g1[5] = (int)(unsigned)(stride0 & 0xFFFFFFFFu);       // tensor_dim0_stride lo
    g1[6] = (int)(unsigned)((stride0 >> 32) & 0xFFFFu);   // stride hi (dim1_stride=0)
    g1[7] = 0;
    i32x4 gz = {0, 0, 0, 0};
#if __clang_major__ >= 23
    i32x8 gz8 = {0, 0, 0, 0, 0, 0, 0, 0};
    __builtin_amdgcn_tensor_load_to_lds(g0, g1, gz, gz, gz8, 0);
#else
    __builtin_amdgcn_tensor_load_to_lds(g0, g1, gz, gz, 0);
#endif
#endif
}

// ---------------------------------------------------------------------------
// fp32 -> bf16 conversion, 4 elements / thread
// ---------------------------------------------------------------------------
__global__ void f32_to_bf16_kernel(const float* __restrict__ in,
                                   __bf16* __restrict__ out, int n4) {
    int i = blockIdx.x * blockDim.x + threadIdx.x;
    if (i >= n4) return;
    float4 v = ((const float4*)in)[i];
    v4bf o;
    o.x = (__bf16)v.x; o.y = (__bf16)v.y; o.z = (__bf16)v.z; o.w = (__bf16)v.w;
    ((v4bf*)out)[i] = o;
}

// ---------------------------------------------------------------------------
// GEMM: C[M,N] = A[M,K] * W[N,K]^T   (bf16 in, WMMA f32 accum)
// Each wave computes a 32(M) x 128(N) tile: 2 A-frags x 8 B-frags -> 16 WMMA
// per 32-K step (20 B128 loads), ~26 FLOP per L2 byte.
// MODE 0: bf16 -> [B,H,S,DH]   (Q / K projection)
// MODE 1: bf16 -> [B,H,DH,S]   (V projection, transposed for P*V B-frags)
// MODE 2: f32  -> [M,N]        (output projection)
// ---------------------------------------------------------------------------
template <int MODE>
__global__ __launch_bounds__(256) void gemm_bf16_kernel(
    const __bf16* __restrict__ A, const __bf16* __restrict__ W,
    void* __restrict__ out, int M, int N, int K) {
    const int lane = threadIdx.x & 31;
    const int ln   = lane & 15;
    const int kh   = lane >> 4;
    const int wave = blockIdx.x * (blockDim.x >> 5) + (threadIdx.x >> 5);
    const int tilesN = N >> 7;
    const int tM = wave / tilesN;
    const int tN = wave % tilesN;
    if (tM * 32 >= M) return;

    const __bf16* arow0 = A + (size_t)(tM * 32 + ln) * K;
    const __bf16* arow1 = arow0 + (size_t)16 * K;

    v8f acc[2][8];
#pragma unroll
    for (int h = 0; h < 2; ++h)
#pragma unroll
        for (int j = 0; j < 8; ++j) acc[h][j] = v8f{};

    for (int k = 0; k < K; k += 32) {
        v16bf a0 = load_frag2(arow0 + k + kh * 8, arow0 + k + 16 + kh * 8);
        v16bf a1 = load_frag2(arow1 + k + kh * 8, arow1 + k + 16 + kh * 8);
#pragma unroll
        for (int j = 0; j < 8; ++j) {
            const int n = tN * 128 + j * 16 + ln;
            const __bf16* wrow = W + (size_t)n * K + k + kh * 8;
            v16bf bf = load_frag2(wrow, wrow + 16);
            acc[0][j] = __builtin_amdgcn_wmma_f32_16x16x32_bf16(
                false, a0, false, bf, (short)0, acc[0][j], false, false);
            acc[1][j] = __builtin_amdgcn_wmma_f32_16x16x32_bf16(
                false, a1, false, bf, (short)0, acc[1][j], false, false);
        }
    }

    // epilogue: C layout -> row = kh*8 + r, col = ln (within each 16x16 tile)
#pragma unroll
    for (int h = 0; h < 2; ++h) {
#pragma unroll
        for (int j = 0; j < 8; ++j) {
            const int n = tN * 128 + j * 16 + ln;
#pragma unroll
            for (int r = 0; r < 8; ++r) {
                const int mm = tM * 32 + h * 16 + kh * 8 + r;
                const float val = acc[h][j][r];
                if (MODE == 2) {
                    ((float*)out)[(size_t)mm * N + n] = val;
                } else {
                    const int b = mm >> 12, s = mm & (SEQ - 1);
                    const int hd = n >> 7, dh = n & (HDIM - 1);
                    const __bf16 bv = (__bf16)val;
                    if (MODE == 0)
                        ((__bf16*)out)[(((size_t)(b * NHEAD + hd) * SEQ + s) * HDIM) + dh] = bv;
                    else
                        ((__bf16*)out)[(((size_t)(b * NHEAD + hd) * HDIM + dh) * SEQ) + s] = bv;
                }
            }
        }
    }
}

// ---------------------------------------------------------------------------
// RoPE applied in-place to Q and K (bf16 [B,H,S,DH]); one thread per (bh,s,pair)
// ---------------------------------------------------------------------------
__global__ void rope_kernel(__bf16* __restrict__ Q, __bf16* __restrict__ Kb) {
    const int i = blockIdx.x * blockDim.x + threadIdx.x;   // < B*H*S*64
    const int dh = i & 63;
    const int s  = (i >> 6) & (SEQ - 1);
    const int bh = i >> 18;
    const size_t base = ((size_t)bh * SEQ + s) * HDIM;

    const float ang = (float)s * __powf(10000.0f, -((float)dh) * (1.0f / 64.0f));
    float sn, cs;
    __sincosf(ang, &sn, &cs);

    float q1 = (float)Q[base + dh], q2 = (float)Q[base + dh + 64];
    Q[base + dh]      = (__bf16)(q1 * cs - q2 * sn);
    Q[base + dh + 64] = (__bf16)(q2 * cs + q1 * sn);

    float k1 = (float)Kb[base + dh], k2 = (float)Kb[base + dh + 64];
    Kb[base + dh]      = (__bf16)(k1 * cs - k2 * sn);
    Kb[base + dh + 64] = (__bf16)(k2 * cs + k1 * sn);
}

// ---------------------------------------------------------------------------
// Flash attention: grid = B*H*(S/128), block = 128 (4 waves, 32 q-rows each).
// K/V tiles are DMA'd into DOUBLE-BUFFERED LDS by the Tensor Data Mover:
// wave 0 issues the TDM for tile i+1 at the top of iteration i (no wait),
// all waves compute tile i from the other buffer, then wave 0 drains
// TENSORcnt and the block barriers -> DMA latency fully overlapped with the
// 32 WMMAs + softmax of the current tile. Each 16 KB tile feeds 128 WMMAs.
// ---------------------------------------------------------------------------
__global__ __launch_bounds__(128) void flash_attn_kernel(
    const __bf16* __restrict__ Q, const __bf16* __restrict__ Kb,
    const __bf16* __restrict__ Vt, __bf16* __restrict__ ctx) {
    __shared__ __bf16 ldsK[2][32 * HDIM];     // [buf][key][dh]   2 x 8 KB
    __shared__ __bf16 ldsV[2][HDIM * 32];     // [buf][dh][key]   2 x 8 KB
    __shared__ __bf16 pbuf[4][16 * 40];       // per-wave P tile, padded rows

    const int lane = threadIdx.x & 31;
    const int ln   = lane & 15;
    const int kh   = lane >> 4;
    const int wave = threadIdx.x >> 5;
    const int bh   = blockIdx.x >> 5;                 // b*H + h
    const int q0   = (blockIdx.x & 31) * 128 + wave * 32;

    const __bf16* Qp  = Q  + (size_t)bh * SEQ * HDIM;
    const __bf16* Kp  = Kb + (size_t)bh * SEQ * HDIM;
    const __bf16* Vtp = Vt + (size_t)bh * HDIM * SEQ;

    // stage key/value tile `tile` (32 keys) into LDS buffer `buf`
    auto stage_tile = [&](int tile, int buf) {
        const int kbase = tile * 32;
#if USE_TDM
        if (wave == 0) {
            // K tile: 32 contiguous rows of 128 bf16 -> 1D copy of 8192 elems
            tdm_load_2d((unsigned)(size_t)(void*)ldsK[buf],
                        Kp + (size_t)kbase * HDIM,
                        8192u, 1u, 8192u, 1u, 8192ull);
            // Vt tile: 128 rows x 32 keys, row stride S
            tdm_load_2d((unsigned)(size_t)(void*)ldsV[buf],
                        Vtp + (size_t)kbase,
                        32u, 128u, 4096u, 128u, (unsigned long long)SEQ);
        }
#else
        {   // cooperative fallback: 128 threads stage 16 KB
            const int tid = threadIdx.x;
            const uint4* src = (const uint4*)(Kp + (size_t)kbase * HDIM);
            uint4* dst = (uint4*)ldsK[buf];
            for (int i = tid; i < 512; i += 128) dst[i] = src[i];
            const uint4* vsrc = (const uint4*)(Vtp + (size_t)tid * SEQ + kbase);
            uint4* vdst = (uint4*)(ldsV[buf] + tid * 32);
#pragma unroll
            for (int c = 0; c < 4; ++c) vdst[c] = vsrc[c];
        }
#endif
    };
    // drain outstanding tile DMA and publish to all waves
    auto publish_tile = [&]() {
#if USE_TDM
        if (wave == 0) __builtin_amdgcn_s_wait_tensorcnt(0);
#endif
        __syncthreads();
    };

    // Q fragments for two 16-row sub-tiles (A-layout, K = head dim)
    v16bf qf[2][4];
#pragma unroll
    for (int t = 0; t < 2; ++t) {
        const __bf16* qrow = Qp + (size_t)(q0 + t * 16 + ln) * HDIM;
#pragma unroll
        for (int dk = 0; dk < 4; ++dk)
            qf[t][dk] = load_frag2(qrow + dk * 32 + kh * 8,
                                   qrow + dk * 32 + 16 + kh * 8);
    }

    float mrow[2][8], lrow[2][8];
    v8f o[2][8];
#pragma unroll
    for (int t = 0; t < 2; ++t)
#pragma unroll
        for (int r = 0; r < 8; ++r) {
            mrow[t][r] = -3.0e38f; lrow[t][r] = 0.0f; o[t][r] = v8f{};
        }

    const float scale = 0.08838834764831843f;         // 1/sqrt(128)
    const int ntiles = SEQ / 32;

    // pipeline prologue: fill buffer 0 with tile 0
    stage_tile(0, 0);
    publish_tile();

    for (int it = 0; it < ntiles; ++it) {
        const int cur = it & 1;
        if (it + 1 < ntiles) stage_tile(it + 1, cur ^ 1);   // async prefetch

        const __bf16* kB = ldsK[cur];
        const __bf16* vB16 = ldsV[cur];

        // ---- scores S = Q K^T : 2 q-tiles x 2 key-tiles, K frags shared ----
        v8f s1[2] = {v8f{}, v8f{}}, s2[2] = {v8f{}, v8f{}};
#pragma unroll
        for (int dk = 0; dk < 4; ++dk) {
            const __bf16* kr1 = kB + ln * HDIM + dk * 32 + kh * 8;
            v16bf b1 = load_frag2(kr1, kr1 + 16);
            const __bf16* kr2 = kB + (16 + ln) * HDIM + dk * 32 + kh * 8;
            v16bf b2 = load_frag2(kr2, kr2 + 16);
#pragma unroll
            for (int t = 0; t < 2; ++t) {
                s1[t] = __builtin_amdgcn_wmma_f32_16x16x32_bf16(
                    false, qf[t][dk], false, b1, (short)0, s1[t], false, false);
                s2[t] = __builtin_amdgcn_wmma_f32_16x16x32_bf16(
                    false, qf[t][dk], false, b2, (short)0, s2[t], false, false);
            }
        }

#pragma unroll
        for (int t = 0; t < 2; ++t) {
            // ---- online softmax (fp32) ----
            float p1[8], p2[8];
#pragma unroll
            for (int r = 0; r < 8; ++r) {
                float a = s1[t][r] * scale, b = s2[t][r] * scale;
                float tm = fmaxf(a, b);
#pragma unroll
                for (int off = 8; off >= 1; off >>= 1)
                    tm = fmaxf(tm, __shfl_xor(tm, off));
                float mnew = fmaxf(mrow[t][r], tm);
                float corr = __expf(mrow[t][r] - mnew);
                p1[r] = __expf(a - mnew);
                p2[r] = __expf(b - mnew);
                float rs = p1[r] + p2[r];
#pragma unroll
                for (int off = 8; off >= 1; off >>= 1)
                    rs += __shfl_xor(rs, off);
                lrow[t][r] = lrow[t][r] * corr + rs;
                mrow[t][r] = mnew;
#pragma unroll
                for (int j = 0; j < 8; ++j) o[t][j][r] *= corr;
            }

            // ---- stage P (C-layout) into LDS, reload as A-layout frag ----
            __bf16* pb = pbuf[wave];
#pragma unroll
            for (int r = 0; r < 8; ++r) {
                const int row = kh * 8 + r;
                pb[row * 40 + ln]      = (__bf16)p1[r];
                pb[row * 40 + 16 + ln] = (__bf16)p2[r];
            }
            const __bf16* prow = pb + ln * 40;
            v16bf pA = load_frag2(prow + kh * 8, prow + 16 + kh * 8);

            // ---- O += P * V from LDS Vt tile ----
#pragma unroll
            for (int j = 0; j < 8; ++j) {
                const __bf16* vr = vB16 + (j * 16 + ln) * 32 + kh * 8;
                v16bf vB = load_frag2(vr, vr + 16);
                o[t][j] = __builtin_amdgcn_wmma_f32_16x16x32_bf16(
                    false, pA, false, vB, (short)0, o[t][j], false, false);
            }
        }

        publish_tile();     // drain next-tile DMA, barrier before buffer swap
    }

    // ---- normalize + store context as [B, S, H*DH] bf16 ----
    const int b = bh >> 4, h = bh & 15;
#pragma unroll
    for (int t = 0; t < 2; ++t)
#pragma unroll
        for (int r = 0; r < 8; ++r) {
            const float inv = 1.0f / lrow[t][r];
            const int s = q0 + t * 16 + kh * 8 + r;
            __bf16* crow = ctx + ((size_t)b * SEQ + s) * DMODEL + h * HDIM;
#pragma unroll
            for (int j = 0; j < 8; ++j)
                crow[j * 16 + ln] = (__bf16)(o[t][j][r] * inv);
        }
}

// ---------------------------------------------------------------------------
// host-side launcher
// ---------------------------------------------------------------------------
extern "C" void kernel_launch(void* const* d_in, const int* in_sizes, int n_in,
                              void* d_out, int out_size, void* d_ws, size_t ws_size,
                              hipStream_t stream) {
    (void)in_sizes; (void)n_in; (void)out_size; (void)ws_size;
    const float* x  = (const float*)d_in[0];
    const float* Wq = (const float*)d_in[1];
    const float* Wk = (const float*)d_in[2];
    const float* Wv = (const float*)d_in[3];
    const float* Wo = (const float*)d_in[4];

    char* ws = (char*)d_ws;
    const size_t XB = (size_t)MTOT * DMODEL * 2;     // 64 MiB
    const size_t WB = (size_t)DMODEL * DMODEL * 2;   //  8 MiB
    __bf16* Xbf = (__bf16*)(ws);
    __bf16* Wqb = (__bf16*)(ws + XB);
    __bf16* Wkb = (__bf16*)(ws + XB + WB);
    __bf16* Wvb = (__bf16*)(ws + XB + 2 * WB);
    __bf16* Wob = (__bf16*)(ws + XB + 3 * WB);
    __bf16* Qb  = (__bf16*)(ws + XB + 4 * WB);
    __bf16* Kb  = (__bf16*)(ws + 2 * XB + 4 * WB);
    __bf16* Vtb = (__bf16*)(ws + 3 * XB + 4 * WB);
    __bf16* Ctx = (__bf16*)(ws + 4 * XB + 4 * WB);

    // fp32 -> bf16 conversions
    {
        int n4 = MTOT * DMODEL / 4;
        f32_to_bf16_kernel<<<(n4 + 255) / 256, 256, 0, stream>>>(x, Xbf, n4);
        int w4 = DMODEL * DMODEL / 4;
        f32_to_bf16_kernel<<<(w4 + 255) / 256, 256, 0, stream>>>(Wq, Wqb, w4);
        f32_to_bf16_kernel<<<(w4 + 255) / 256, 256, 0, stream>>>(Wk, Wkb, w4);
        f32_to_bf16_kernel<<<(w4 + 255) / 256, 256, 0, stream>>>(Wv, Wvb, w4);
        f32_to_bf16_kernel<<<(w4 + 255) / 256, 256, 0, stream>>>(Wo, Wob, w4);
    }

    // projections: (M/32)*(N/128) waves, 8 waves per 256-thread block
    const int gemm_blocks = (MTOT / 32) * (DMODEL / 128) / 8;   // 1024
    gemm_bf16_kernel<0><<<gemm_blocks, 256, 0, stream>>>(Xbf, Wqb, Qb,  MTOT, DMODEL, DMODEL);
    gemm_bf16_kernel<0><<<gemm_blocks, 256, 0, stream>>>(Xbf, Wkb, Kb,  MTOT, DMODEL, DMODEL);
    gemm_bf16_kernel<1><<<gemm_blocks, 256, 0, stream>>>(Xbf, Wvb, Vtb, MTOT, DMODEL, DMODEL);

    // RoPE on Q and K
    {
        int n = BATCH * NHEAD * SEQ * 64;                       // 16.7M threads
        rope_kernel<<<n / 256, 256, 0, stream>>>(Qb, Kb);
    }

    // flash attention: B*H*(S/128) blocks of 128 threads
    flash_attn_kernel<<<BATCH * NHEAD * (SEQ / 128), 128, 0, stream>>>(Qb, Kb, Vtb, Ctx);

    // output projection -> fp32
    gemm_bf16_kernel<2><<<gemm_blocks, 256, 0, stream>>>(Ctx, Wob, d_out, MTOT, DMODEL, DMODEL);
}
